// FragGNN_36163624632848
// MI455X (gfx1250) — compile-verified, matrix operand order checked by reading, not run on hardware
//
#include <hip/hip_runtime.h>
#include <hip/hip_bf16.h>

// ---------------------------------------------------------------------------
// FragGNN forward for MI455X (gfx1250, wave32, WMMA f16->f32 16x16x32)
// ---------------------------------------------------------------------------

#define H_DIM   256
#define NF_DIM  128
#define L_NUM   2
#define B_NUM   8192
#define NPG_NUM 25
#define N_NODES (B_NUM * NPG_NUM)     // 204800
#define E_NUM   (4 * N_NODES)         // 819200
#define NROOT   2048
#define FP_DIM  4096
#define K1_PAD  800                   // 781 padded to 25*32

typedef __attribute__((ext_vector_type(16))) _Float16 v16h;
typedef __attribute__((ext_vector_type(8)))  float    v8f;

// Load one 16x32-f16 operand fragment for lane (row/col = lane%16).
// p points at the 32-element K-run for this lane's row (A) or column (B^T).
// Per-lane K packing per CDNA5 ISA 7.12.2:
//   lanes 0-15 : VGPR0-3 -> K 0..7,  VGPR4-7 -> K 16..23
//   lanes 16-31: VGPR0-3 -> K 8..15, VGPR4-7 -> K 24..31
__device__ __forceinline__ v16h frag_ld(const _Float16* __restrict__ p, int hi) {
  v16h f;
  const int kb = hi ? 8 : 0;
#pragma unroll
  for (int j = 0; j < 8; ++j) {
    const int k0 = (j < 4) ? (kb + 2 * j) : (16 + kb + 2 * (j - 4));
    f[2 * j]     = p[k0];
    f[2 * j + 1] = p[k0 + 1];
  }
  return f;
}

__device__ __forceinline__ v8f wmma16(v16h a, v16h b, v8f c) {
  return __builtin_amdgcn_wmma_f32_16x16x32_f16(false, a, false, b,
                                                (short)0, c, false, false);
}

__device__ __forceinline__ float sigm(float x) {
  return 1.0f / (1.0f + __expf(-x));
}

// ---------------------------------------------------------------------------
// Prep: transpose fp32 weight [K x Nc] -> f16 [Nc x Kpad] (zero pad K tail)
// ---------------------------------------------------------------------------
__global__ void k_transpose_f16(const float* __restrict__ src, _Float16* __restrict__ dst,
                                int K, int Nc, int Kpad) {
  long idx = (long)blockIdx.x * blockDim.x + threadIdx.x;
  long total = (long)Nc * Kpad;
  if (idx >= total) return;
  int n = (int)(idx / Kpad);
  int k = (int)(idx % Kpad);
  dst[idx] = (k < K) ? (_Float16)src[(long)k * Nc + n] : (_Float16)0.0f;
}

// ---------------------------------------------------------------------------
// Generic GEMM: C[M x 256] = opt_relu(A[M x K] @ W[K x 256] + bias)
// Wt is the f16 transposed copy [256 x K]. One WG = 16 rows, 8 waves x 2 tiles.
// ---------------------------------------------------------------------------
__global__ __launch_bounds__(256)
void k_gemm_nc256(const float* __restrict__ A, int K,
                  const _Float16* __restrict__ Wt,
                  const float* __restrict__ bias,
                  float* __restrict__ C, int relu) {
  __shared__ _Float16 As[16 * 32];
  const int tid  = threadIdx.x;
  const int lane = tid & 31, wave = tid >> 5;
  const int hi = lane >> 4, ln = lane & 15;
  const long row0 = (long)blockIdx.x * 16;
  const int c0 = wave * 32;

  v8f acc0 = {}, acc1 = {};
  for (int kb = 0; kb < K; kb += 32) {
    __syncthreads();
    for (int i = tid; i < 512; i += 256) {
      const int r = i >> 5, k = i & 31;
      As[i] = (_Float16)A[(row0 + r) * (long)K + kb + k];
    }
    __syncthreads();
    const v16h a  = frag_ld(&As[ln * 32], hi);
    const v16h b0 = frag_ld(&Wt[(long)(c0 + ln)      * K + kb], hi);
    const v16h b1 = frag_ld(&Wt[(long)(c0 + 16 + ln) * K + kb], hi);
    acc0 = wmma16(a, b0, acc0);
    acc1 = wmma16(a, b1, acc1);
  }
#pragma unroll
  for (int t = 0; t < 2; ++t) {
    const v8f& acc = t ? acc1 : acc0;
    const int c = c0 + t * 16 + ln;
    const float bv = bias[c];
#pragma unroll
    for (int v = 0; v < 8; ++v) {
      const int r = v + 8 * hi;
      float x = acc[v] + bv;
      if (relu) x = fmaxf(x, 0.0f);
      C[(row0 + r) * H_DIM + c] = x;
    }
  }
}

// ---------------------------------------------------------------------------
// Zero fill (float4)
// ---------------------------------------------------------------------------
__global__ void k_zero(float4* __restrict__ p, long n4) {
  long i = (long)blockIdx.x * blockDim.x + threadIdx.x;
  if (i < n4) p[i] = make_float4(0.f, 0.f, 0.f, 0.f);
}

// ---------------------------------------------------------------------------
// Edge scatter: a[dst] += m[src]  (fully random; float atomics)
// One thread = (edge, 4-float chunk)
// ---------------------------------------------------------------------------
__global__ void k_scatter(const float* __restrict__ M,
                          const int* __restrict__ esrc,
                          const int* __restrict__ edst,
                          float* __restrict__ Aagg) {
  const long idx = (long)blockIdx.x * blockDim.x + threadIdx.x;
  if (idx >= (long)E_NUM * 64) return;
  const int e = (int)(idx >> 6);
  const int c = (int)(idx & 63) << 2;
  const long s = (long)esrc[e] * H_DIM + c;
  const long d = (long)edst[e] * H_DIM + c;
  const float4 v = *(const float4*)(M + s);
  atomicAdd(Aagg + d + 0, v.x);
  atomicAdd(Aagg + d + 1, v.y);
  atomicAdd(Aagg + d + 2, v.z);
  atomicAdd(Aagg + d + 3, v.w);
}

// ---------------------------------------------------------------------------
// Fused GatedGraphConv GRU cell: h = GRU(a, h), in place (rows disjoint).
// gi = a @ W_ih + b_ih, gh = h @ W_hh + b_hh, both [16 x 768] computed with
// 6 WMMA accumulators per wave col-tile; gate math stays in registers.
// ---------------------------------------------------------------------------
__global__ __launch_bounds__(256)
void k_gru(float* __restrict__ Hst,
           const float* __restrict__ Aagg,
           const _Float16* __restrict__ WihT,   // [768 x 256] f16
           const _Float16* __restrict__ WhhT,   // [768 x 256] f16
           const float* __restrict__ b_ih,      // [768]
           const float* __restrict__ b_hh) {    // [768]
  __shared__ float    Hs[16 * H_DIM];
  __shared__ _Float16 Hh[16 * H_DIM];
  __shared__ _Float16 Ah[16 * H_DIM];
  const int tid  = threadIdx.x;
  const int lane = tid & 31, wave = tid >> 5;
  const int hi = lane >> 4, ln = lane & 15;
  const long row0 = (long)blockIdx.x * 16;

  for (int i = tid; i < 16 * H_DIM; i += 256) {
    const int r = i >> 8, k = i & 255;
    const float hv = Hst[(row0 + r) * H_DIM + k];
    const float av = Aagg[(row0 + r) * H_DIM + k];
    Hs[i] = hv;
    Hh[i] = (_Float16)hv;
    Ah[i] = (_Float16)av;
  }
  __syncthreads();

  for (int t = 0; t < 2; ++t) {
    const int c = wave * 32 + t * 16;
    v8f ir = {}, iz = {}, inn = {}, hr = {}, hz = {}, hn = {};
#pragma unroll
    for (int kc = 0; kc < 8; ++kc) {
      const int kb = kc * 32;
      const v16h af = frag_ld(&Ah[ln * H_DIM + kb], hi);
      const v16h hf = frag_ld(&Hh[ln * H_DIM + kb], hi);
      const v16h br = frag_ld(&WihT[(long)(c + ln)       * H_DIM + kb], hi);
      const v16h bz = frag_ld(&WihT[(long)(c + 256 + ln) * H_DIM + kb], hi);
      const v16h bn = frag_ld(&WihT[(long)(c + 512 + ln) * H_DIM + kb], hi);
      ir  = wmma16(af, br, ir);
      iz  = wmma16(af, bz, iz);
      inn = wmma16(af, bn, inn);
      const v16h cr = frag_ld(&WhhT[(long)(c + ln)       * H_DIM + kb], hi);
      const v16h cz = frag_ld(&WhhT[(long)(c + 256 + ln) * H_DIM + kb], hi);
      const v16h cn = frag_ld(&WhhT[(long)(c + 512 + ln) * H_DIM + kb], hi);
      hr = wmma16(hf, cr, hr);
      hz = wmma16(hf, cz, hz);
      hn = wmma16(hf, cn, hn);
    }
    const int cc = c + ln;
    const float bir = b_ih[cc], biz = b_ih[cc + 256], bin = b_ih[cc + 512];
    const float bhr = b_hh[cc], bhz = b_hh[cc + 256], bhn = b_hh[cc + 512];
#pragma unroll
    for (int v = 0; v < 8; ++v) {
      const int r = v + 8 * hi;
      const float rg = sigm(ir[v] + bir + hr[v] + bhr);
      const float zg = sigm(iz[v] + biz + hz[v] + bhz);
      const float ng = tanhf(inn[v] + bin + rg * (hn[v] + bhn));
      const float ho = Hs[r * H_DIM + cc];
      Hst[(row0 + r) * H_DIM + cc] = (1.0f - zg) * ng + zg * ho;
    }
  }
}

// ---------------------------------------------------------------------------
// Per-graph mean over fixed NPG nodes
// ---------------------------------------------------------------------------
__global__ void k_pool(const float* __restrict__ Hst, float* __restrict__ Avg) {
  const long idx = (long)blockIdx.x * blockDim.x + threadIdx.x;
  if (idx >= (long)B_NUM * H_DIM) return;
  const int g = (int)(idx >> 8);
  const int c = (int)(idx & 255);
  const long base = (long)g * NPG_NUM * H_DIM + c;
  float s = 0.0f;
#pragma unroll
  for (int i = 0; i < NPG_NUM; ++i) s += Hst[base + (long)i * H_DIM];
  Avg[idx] = s * (1.0f / (float)NPG_NUM);
}

// ---------------------------------------------------------------------------
// Fused output head: build cat row (800 wide, f16) in LDS, MLP1+ReLU,
// MLP2+ReLU fused with out_W dot, sigmoid.
// ---------------------------------------------------------------------------
__global__ __launch_bounds__(256)
void k_final(const float* __restrict__ Hst, const float* __restrict__ RootE,
             const float* __restrict__ Avg, const int* __restrict__ ind_maps,
             const int* __restrict__ broken, const int* __restrict__ node2graph,
             const _Float16* __restrict__ W1T,   // [256 x 800] f16
             const float* __restrict__ b1,
             const _Float16* __restrict__ W2T,   // [256 x 256] f16
             const float* __restrict__ b2,
             const float* __restrict__ outW, const float* __restrict__ outb,
             float* __restrict__ out) {
  __shared__ _Float16 Acat[16 * K1_PAD];
  __shared__ _Float16 H1[16 * H_DIM];
  __shared__ float oacc[16];
  const int tid  = threadIdx.x;
  const int lane = tid & 31, wave = tid >> 5;
  const int hi = lane >> 4, ln = lane & 15;
  const long n0 = (long)blockIdx.x * 16;

  if (tid < 16) oacc[tid] = 0.0f;
  for (int i = tid; i < 16 * K1_PAD; i += 256) {
    const int r = i / K1_PAD, c = i % K1_PAD;
    const long n = n0 + r;
    const int g = node2graph[n];
    float v;
    if (c < 256) {
      v = RootE[(long)ind_maps[g] * H_DIM + c];
    } else if (c < 512) {
      const int cc = c - 256;
      v = RootE[(long)ind_maps[g] * H_DIM + cc] - Avg[(long)g * H_DIM + cc];
    } else if (c < 768) {
      v = Hst[n * H_DIM + (c - 512)];
    } else if (c < 781) {
      v = ((c - 768) == broken[g]) ? 1.0f : 0.0f;
    } else {
      v = 0.0f;
    }
    Acat[i] = (_Float16)v;
  }
  __syncthreads();

  // MLP layer 1: [16 x 800] @ [800 x 256] + b1, ReLU -> H1 (f16 LDS)
  for (int t = 0; t < 2; ++t) {
    const int c = wave * 32 + t * 16;
    v8f acc = {};
    for (int kc = 0; kc < K1_PAD / 32; ++kc) {
      const int kb = kc * 32;
      const v16h a = frag_ld(&Acat[ln * K1_PAD + kb], hi);
      const v16h b = frag_ld(&W1T[(long)(c + ln) * K1_PAD + kb], hi);
      acc = wmma16(a, b, acc);
    }
    const int cc = c + ln;
    const float bv = b1[cc];
#pragma unroll
    for (int v = 0; v < 8; ++v) {
      const int r = v + 8 * hi;
      H1[r * H_DIM + cc] = (_Float16)fmaxf(acc[v] + bv, 0.0f);
    }
  }
  __syncthreads();

  // MLP layer 2 + ReLU + dot(out_W) reduced through LDS
  for (int t = 0; t < 2; ++t) {
    const int c = wave * 32 + t * 16;
    v8f acc = {};
#pragma unroll
    for (int kc = 0; kc < 8; ++kc) {
      const int kb = kc * 32;
      const v16h a = frag_ld(&H1[ln * H_DIM + kb], hi);
      const v16h b = frag_ld(&W2T[(long)(c + ln) * H_DIM + kb], hi);
      acc = wmma16(a, b, acc);
    }
    const int cc = c + ln;
    const float bv = b2[cc], wv = outW[cc];
#pragma unroll
    for (int v = 0; v < 8; ++v) {
      const int r = v + 8 * hi;
      const float h2 = fmaxf(acc[v] + bv, 0.0f);
      atomicAdd(&oacc[r], h2 * wv);
    }
  }
  __syncthreads();
  if (tid < 16) {
    out[n0 + tid] = sigm(oacc[tid] + outb[0]);
  }
}

// ---------------------------------------------------------------------------
// Host launcher
// ---------------------------------------------------------------------------
extern "C" void kernel_launch(void* const* d_in, const int* in_sizes, int n_in,
                              void* d_out, int out_size, void* d_ws, size_t ws_size,
                              hipStream_t stream) {
  (void)in_sizes; (void)n_in; (void)out_size; (void)ws_size;

  const float* node_x   = (const float*)d_in[0];
  const float* root_fp  = (const float*)d_in[1];
  const int*   edge_src = (const int*)d_in[2];
  const int*   edge_dst = (const int*)d_in[3];
  const int*   ind_maps = (const int*)d_in[4];
  const int*   broken   = (const int*)d_in[5];
  const int*   node2g   = (const int*)d_in[6];
  const float* root_W   = (const float*)d_in[7];
  const float* root_b   = (const float*)d_in[8];
  const float* nin_W    = (const float*)d_in[9];
  const float* nin_b    = (const float*)d_in[10];
  const float* edge_W   = (const float*)d_in[11];
  const float* edge_b   = (const float*)d_in[12];
  const float* W_ih     = (const float*)d_in[13];
  const float* b_ih     = (const float*)d_in[14];
  const float* W_hh     = (const float*)d_in[15];
  const float* b_hh     = (const float*)d_in[16];
  const float* mlp_W1   = (const float*)d_in[17];
  const float* mlp_b1   = (const float*)d_in[18];
  const float* mlp_W2   = (const float*)d_in[19];
  const float* mlp_b2   = (const float*)d_in[20];
  const float* out_W    = (const float*)d_in[21];
  const float* out_b    = (const float*)d_in[22];
  float* out = (float*)d_out;

  // Workspace bump allocator (256B aligned)
  char* wp = (char*)d_ws;
  auto alloc = [&](size_t bytes) -> void* {
    void* p = (void*)wp;
    wp += (bytes + 255) & ~(size_t)255;
    return p;
  };
  _Float16* rootWt = (_Float16*)alloc((size_t)256 * FP_DIM * 2);
  _Float16* ninWt  = (_Float16*)alloc((size_t)256 * NF_DIM * 2);
  _Float16* edgeWt = (_Float16*)alloc((size_t)L_NUM * 256 * 256 * 2);
  _Float16* WihT   = (_Float16*)alloc((size_t)L_NUM * 768 * 256 * 2);
  _Float16* WhhT   = (_Float16*)alloc((size_t)L_NUM * 768 * 256 * 2);
  _Float16* W1T    = (_Float16*)alloc((size_t)256 * K1_PAD * 2);
  _Float16* W2T    = (_Float16*)alloc((size_t)256 * 256 * 2);
  float* rootEmb = (float*)alloc((size_t)NROOT * H_DIM * 4);
  float* fragH   = (float*)alloc((size_t)N_NODES * H_DIM * 4);
  float* Mbuf    = (float*)alloc((size_t)N_NODES * H_DIM * 4);
  float* Abuf    = (float*)alloc((size_t)N_NODES * H_DIM * 4);
  float* Avg     = (float*)alloc((size_t)B_NUM * H_DIM * 4);

  const int TPB = 256;
  auto gb = [](long n, int t) { return (int)((n + t - 1) / t); };

  // ---- weight prep (fp32 [KxNc] -> f16 [NcxKpad]) ----
  k_transpose_f16<<<gb((long)256 * FP_DIM, TPB), TPB, 0, stream>>>(root_W, rootWt, FP_DIM, 256, FP_DIM);
  k_transpose_f16<<<gb((long)256 * NF_DIM, TPB), TPB, 0, stream>>>(nin_W, ninWt, NF_DIM, 256, NF_DIM);
  for (int l = 0; l < L_NUM; ++l) {
    k_transpose_f16<<<gb(256L * 256, TPB), TPB, 0, stream>>>(
        edge_W + (size_t)l * 256 * 256, edgeWt + (size_t)l * 256 * 256, 256, 256, 256);
    k_transpose_f16<<<gb(768L * 256, TPB), TPB, 0, stream>>>(
        W_ih + (size_t)l * 256 * 768, WihT + (size_t)l * 768 * 256, 256, 768, 256);
    k_transpose_f16<<<gb(768L * 256, TPB), TPB, 0, stream>>>(
        W_hh + (size_t)l * 256 * 768, WhhT + (size_t)l * 768 * 256, 256, 768, 256);
  }
  k_transpose_f16<<<gb((long)256 * K1_PAD, TPB), TPB, 0, stream>>>(mlp_W1, W1T, 781, 256, K1_PAD);
  k_transpose_f16<<<gb(256L * 256, TPB), TPB, 0, stream>>>(mlp_W2, W2T, 256, 256, 256);

  // ---- root encoder: relu(root_fp @ root_W + b) ----
  k_gemm_nc256<<<NROOT / 16, TPB, 0, stream>>>(root_fp, FP_DIM, rootWt, root_b, rootEmb, 1);

  // ---- node embedding: node_x @ node_in_W + b ----
  k_gemm_nc256<<<N_NODES / 16, TPB, 0, stream>>>(node_x, NF_DIM, ninWt, nin_b, fragH, 0);

  // ---- 2 GatedGraphConv layers ----
  for (int l = 0; l < L_NUM; ++l) {
    k_gemm_nc256<<<N_NODES / 16, TPB, 0, stream>>>(
        fragH, 256, edgeWt + (size_t)l * 256 * 256, edge_b + (size_t)l * 256, Mbuf, 0);
    k_zero<<<gb((long)N_NODES * H_DIM / 4, TPB), TPB, 0, stream>>>(
        (float4*)Abuf, (long)N_NODES * H_DIM / 4);
    k_scatter<<<gb((long)E_NUM * 64, TPB), TPB, 0, stream>>>(Mbuf, edge_src, edge_dst, Abuf);
    k_gru<<<N_NODES / 16, TPB, 0, stream>>>(
        fragH, Abuf,
        WihT + (size_t)l * 768 * 256, WhhT + (size_t)l * 768 * 256,
        b_ih + (size_t)l * 768, b_hh + (size_t)l * 768);
  }

  // ---- per-graph mean pool ----
  k_pool<<<gb((long)B_NUM * H_DIM, TPB), TPB, 0, stream>>>(fragH, Avg);

  // ---- fused output head ----
  k_final<<<N_NODES / 16, TPB, 0, stream>>>(
      fragH, rootEmb, Avg, ind_maps, broken, node2g,
      W1T, mlp_b1, W2T, mlp_b2, out_W, out_b, out);
}
